// ConTrustModel_81793357185436
// MI455X (gfx1250) — matrix operand order repaired and checked by reading, not compiled, not run on hardware
//
#include <hip/hip_runtime.h>

// Problem constants (from reference): B=4, S=2048, D=512, H=8, HD=64
#define B_   4
#define S_   2048
#define D_   512
#define H_   8
#define HD_  64
#define BH_  (B_ * H_)
#define M_   (B_ * S_)   // 8192 rows for all [B*S, D] GEMMs

typedef __attribute__((ext_vector_type(16))) _Float16 v16h;
typedef __attribute__((ext_vector_type(8)))  _Float16 v8h;
typedef __attribute__((ext_vector_type(8)))  float    v8f;
typedef __attribute__((ext_vector_type(4)))  float    v4f;
typedef int v4i_gcc __attribute__((__vector_size__(16)));  // matches builtin param

#define HAVE_ASYNC_LDS __has_builtin(__builtin_amdgcn_global_load_async_to_lds_b128)

// 16B global->LDS copy: async (ASYNCcnt-tracked, no VGPR round trip) when the
// gfx1250 builtin exists, else a plain 16B copy.
__device__ __forceinline__ void copy16_g2l(_Float16* lds_dst, const _Float16* gsrc) {
#if HAVE_ASYNC_LDS
  __builtin_amdgcn_global_load_async_to_lds_b128((v4i_gcc*)gsrc, (v4i_gcc*)lds_dst,
                                                 0, 0);
#else
  *(v8h*)lds_dst = *(const v8h*)gsrc;
#endif
}
__device__ __forceinline__ void async_join() {
#if HAVE_ASYNC_LDS
  asm volatile("s_wait_asynccnt 0" ::: "memory");
#endif
}

__device__ __forceinline__ v8f wmma_f16(v16h a, v16h b, v8f c) {
  return __builtin_amdgcn_wmma_f32_16x16x32_f16(false, a, false, b, (short)0, c,
                                                false, false);
}

__device__ __forceinline__ v16h frag2(const _Float16* p0, const _Float16* p1) {
  v16h r;
  *reinterpret_cast<v8h*>(&r)       = *reinterpret_cast<const v8h*>(p0);
  *(reinterpret_cast<v8h*>(&r) + 1) = *reinterpret_cast<const v8h*>(p1);
  return r;
}

// 16 contiguous f32 -> 16 f16 in LDS via packed regs (2x ds_store_b128).
__device__ __forceinline__ void cvt16_f32_to_lds(_Float16* dst, const float* src) {
  const v4f* s4 = (const v4f*)src;
  v4f f0 = s4[0], f1 = s4[1], f2 = s4[2], f3 = s4[3];
  v8h h0, h1;
#pragma unroll
  for (int i = 0; i < 4; ++i) {
    h0[i]     = (_Float16)f0[i];
    h0[4 + i] = (_Float16)f1[i];
    h1[i]     = (_Float16)f2[i];
    h1[4 + i] = (_Float16)f3[i];
  }
  ((v8h*)dst)[0] = h0;
  ((v8h*)dst)[1] = h1;
}

__device__ __forceinline__ float sigmoidf_(float x) {
  return 1.0f / (1.0f + __expf(-x));
}

// ---------------------------------------------------------------------------
// 1) QKV projection: C = X(MxK)@W(KxN)+bias -> f16, head layout via strides:
//    dst = (b*H+h)*S*HD + s*sstride + hd*hstride
//    (Q,K: sstride=HD,hstride=1 -> [B,H,S,HD];  V: sstride=1,hstride=S -> [B,H,HD,S])
// ---------------------------------------------------------------------------
__global__ __launch_bounds__(256) void proj_gemm_kernel(
    const float* __restrict__ X, const float* __restrict__ W,
    const float* __restrict__ bias, _Float16* __restrict__ Out,
    int sstride, int hstride) {
  const int K = D_, N = D_;
  __shared__ _Float16 As[128 * 32];  // [m][k]
  __shared__ _Float16 Bs[64 * 32];   // [n][k]
  const int tile_m = blockIdx.x * 128;
  const int tile_n = blockIdx.y * 64;
  const int tid = threadIdx.x, lane = tid & 31, wave = tid >> 5;

  v8f acc[4] = {};
  const int arow = tid >> 1, ac0 = (tid & 1) * 16;  // A: 2 thr/row, 16 f32 each
  const int bn = tid & 63, bk0 = (tid >> 6) * 8;    // B: 1 col/thr, 8 k's each

  for (int k0 = 0; k0 < K; k0 += 32) {
    cvt16_f32_to_lds(&As[arow * 32 + ac0],
                     X + (size_t)(tile_m + arow) * K + k0 + ac0);
    {
      const float* wg = W + (size_t)(k0 + bk0) * N + tile_n + bn;
      v8h h;
#pragma unroll
      for (int i = 0; i < 8; ++i) h[i] = (_Float16)wg[(size_t)i * N];
      *(v8h*)&Bs[bn * 32 + bk0] = h;
    }
    __syncthreads();

    const int ak = (lane < 16) ? 0 : 8;
    const int mr = wave * 16 + (lane & 15);
    v16h a = frag2(&As[mr * 32 + ak], &As[mr * 32 + ak + 16]);
    const int bk = (lane < 16) ? 0 : 8;
#pragma unroll
    for (int ns = 0; ns < 4; ++ns) {
      const int nr = ns * 16 + (lane & 15);
      v16h b = frag2(&Bs[nr * 32 + bk], &Bs[nr * 32 + bk + 16]);
      acc[ns] = wmma_f16(a, b, acc[ns]);
    }
    __syncthreads();
  }

  const int roff = (lane < 16) ? 0 : 8;
#pragma unroll
  for (int ns = 0; ns < 4; ++ns) {
    const int n = tile_n + ns * 16 + (lane & 15);
    const int h = n >> 6, hd = n & 63;
#pragma unroll
    for (int r = 0; r < 8; ++r) {
      const int m = tile_m + wave * 16 + roff + r;
      const int b = m / S_, s = m % S_;
      const float v = acc[ns][r] + bias[n];
      Out[(size_t)(b * H_ + h) * S_ * HD_ + (size_t)s * sstride +
          (size_t)hd * hstride] = (_Float16)v;
    }
  }
}

// ---------------------------------------------------------------------------
// 2) Trust projection: tp[b,s,hd] = trust[b,s]*Wt[hd] + bt[hd]  (f16)
// ---------------------------------------------------------------------------
__global__ void trust_proj_kernel(const float* __restrict__ trust,
                                  const float* __restrict__ Wt,
                                  const float* __restrict__ bt,
                                  _Float16* __restrict__ tp) {
  const int i = blockIdx.x * 256 + threadIdx.x;
  if (i >= M_ * HD_) return;
  const int hd = i & (HD_ - 1), bs = i >> 6;
  tp[i] = (_Float16)(trust[bs] * Wt[hd] + bt[hd]);
}

// ---------------------------------------------------------------------------
// 3) Scores: attn_raw = Q@K^T/8 + 0.1*sigmoid(Tp@Tp^T); async-staged f16 tiles.
// ---------------------------------------------------------------------------
__global__ __launch_bounds__(256) void scores_kernel(
    const _Float16* __restrict__ qh, const _Float16* __restrict__ kh,
    const _Float16* __restrict__ tp, float* __restrict__ attn) {
  __shared__ _Float16 Qs[128 * 64];
  __shared__ _Float16 Tq[128 * 64];
  __shared__ _Float16 Ks[64 * 64];
  __shared__ _Float16 Tk[64 * 64];
  const int bh = blockIdx.z, b = bh >> 3;
  const int tile_q = blockIdx.x * 128;
  const int tile_k = blockIdx.y * 64;
  const int tid = threadIdx.x, lane = tid & 31, wave = tid >> 5;

  {  // Q/Tq: 128 rows x 64 halves; 2 thr/row, 32 halves each (4x16B)
    const int row = tid >> 1, c0 = (tid & 1) * 32;
    const _Float16* gq = qh + ((size_t)bh * S_ + tile_q + row) * HD_ + c0;
    const _Float16* gt = tp + ((size_t)b * S_ + tile_q + row) * HD_ + c0;
#pragma unroll
    for (int j = 0; j < 4; ++j) {
      copy16_g2l(&Qs[row * 64 + c0 + j * 8], gq + j * 8);
      copy16_g2l(&Tq[row * 64 + c0 + j * 8], gt + j * 8);
    }
  }
  {  // K/Tk: 64 rows x 64 halves; 4 thr/row, 16 halves each (2x16B)
    const int row = tid >> 2, c0 = (tid & 3) * 16;
    const _Float16* gk = kh + ((size_t)bh * S_ + tile_k + row) * HD_ + c0;
    const _Float16* gt = tp + ((size_t)b * S_ + tile_k + row) * HD_ + c0;
#pragma unroll
    for (int j = 0; j < 2; ++j) {
      copy16_g2l(&Ks[row * 64 + c0 + j * 8], gk + j * 8);
      copy16_g2l(&Tk[row * 64 + c0 + j * 8], gt + j * 8);
    }
  }
  async_join();
  __syncthreads();

  v8f accS[4] = {};
  v8f accT[4] = {};
#pragma unroll
  for (int k0 = 0; k0 < HD_; k0 += 32) {
    const int ak = k0 + ((lane < 16) ? 0 : 8);
    const int mr = wave * 16 + (lane & 15);
    v16h aq = frag2(&Qs[mr * 64 + ak], &Qs[mr * 64 + ak + 16]);
    v16h at = frag2(&Tq[mr * 64 + ak], &Tq[mr * 64 + ak + 16]);
    const int bk = k0 + ((lane < 16) ? 0 : 8);
#pragma unroll
    for (int ns = 0; ns < 4; ++ns) {
      const int nr = ns * 16 + (lane & 15);
      v16h bq = frag2(&Ks[nr * 64 + bk], &Ks[nr * 64 + bk + 16]);
      v16h bt = frag2(&Tk[nr * 64 + bk], &Tk[nr * 64 + bk + 16]);
      accS[ns] = wmma_f16(aq, bq, accS[ns]);
      accT[ns] = wmma_f16(at, bt, accT[ns]);
    }
  }

  const int roff = (lane < 16) ? 0 : 8;
#pragma unroll
  for (int ns = 0; ns < 4; ++ns) {
    const int kk = tile_k + ns * 16 + (lane & 15);
#pragma unroll
    for (int r = 0; r < 8; ++r) {
      const int q = tile_q + wave * 16 + roff + r;
      const float sc = accS[ns][r] * 0.125f + 0.1f * sigmoidf_(accT[ns][r]);
      __builtin_nontemporal_store(sc, &attn[((size_t)bh * S_ + q) * S_ + kk]);
    }
  }
}

// ---------------------------------------------------------------------------
// 4) Row softmax in-place (NT vector I/O). One WG per row.
// ---------------------------------------------------------------------------
__global__ __launch_bounds__(256) void softmax_kernel(float* __restrict__ attn) {
  __shared__ float red[256];
  v4f* p4 = (v4f*)(attn + (size_t)blockIdx.x * S_);
  const int tid = threadIdx.x;
  v4f a = __builtin_nontemporal_load(p4 + tid * 2);
  v4f c = __builtin_nontemporal_load(p4 + tid * 2 + 1);
  float m = -3.0e38f;
#pragma unroll
  for (int i = 0; i < 4; ++i) m = fmaxf(m, fmaxf(a[i], c[i]));
  red[tid] = m;
  __syncthreads();
  for (int s = 128; s > 0; s >>= 1) {
    if (tid < s) red[tid] = fmaxf(red[tid], red[tid + s]);
    __syncthreads();
  }
  m = red[0];
  __syncthreads();
  float sum = 0.0f;
#pragma unroll
  for (int i = 0; i < 4; ++i) {
    a[i] = __expf(a[i] - m);
    c[i] = __expf(c[i] - m);
    sum += a[i] + c[i];
  }
  red[tid] = sum;
  __syncthreads();
  for (int s = 128; s > 0; s >>= 1) {
    if (tid < s) red[tid] += red[tid + s];
    __syncthreads();
  }
  const float inv = 1.0f / red[0];
#pragma unroll
  for (int i = 0; i < 4; ++i) { a[i] *= inv; c[i] *= inv; }
  __builtin_nontemporal_store(a, p4 + tid * 2);
  __builtin_nontemporal_store(c, p4 + tid * 2 + 1);
}

// ---------------------------------------------------------------------------
// 5) attended = attn @ V. V is pre-transposed [B,H,HD,S] so the B tile is a
//    pure contiguous f16 async copy; attn fp32->f16 converts in parallel.
// ---------------------------------------------------------------------------
__global__ __launch_bounds__(256) void av_gemm_kernel(
    const float* __restrict__ attn, const _Float16* __restrict__ vhT,
    float* __restrict__ a32, _Float16* __restrict__ a16) {
  __shared__ _Float16 Ps[128 * 32];  // [q][k]
  __shared__ _Float16 Vs[64 * 32];   // [hd][k]
  const int bh = blockIdx.y, b = bh >> 3, h = bh & 7;
  const int tile_q = blockIdx.x * 128;
  const int tid = threadIdx.x, lane = tid & 31, wave = tid >> 5;

  v8f acc[4] = {};
  const int prow = tid >> 1, pc0 = (tid & 1) * 16;
  const int vn = tid >> 2, vc0 = (tid & 3) * 8;

  for (int k0 = 0; k0 < S_; k0 += 32) {
    copy16_g2l(&Vs[vn * 32 + vc0],
               vhT + ((size_t)bh * HD_ + vn) * S_ + k0 + vc0);
    cvt16_f32_to_lds(&Ps[prow * 32 + pc0],
                     attn + ((size_t)bh * S_ + tile_q + prow) * S_ + k0 + pc0);
    async_join();
    __syncthreads();

    const int ak = (lane < 16) ? 0 : 8;
    const int mr = wave * 16 + (lane & 15);
    v16h a = frag2(&Ps[mr * 32 + ak], &Ps[mr * 32 + ak + 16]);
    const int bk = (lane < 16) ? 0 : 8;
#pragma unroll
    for (int ns = 0; ns < 4; ++ns) {
      const int nr = ns * 16 + (lane & 15);
      v16h bb = frag2(&Vs[nr * 32 + bk], &Vs[nr * 32 + bk + 16]);
      acc[ns] = wmma_f16(a, bb, acc[ns]);
    }
    __syncthreads();
  }

  const int roff = (lane < 16) ? 0 : 8;
#pragma unroll
  for (int ns = 0; ns < 4; ++ns) {
    const int hd = ns * 16 + (lane & 15);
#pragma unroll
    for (int r = 0; r < 8; ++r) {
      const int q = tile_q + wave * 16 + roff + r;
      const size_t idx = ((size_t)b * S_ + q) * D_ + h * HD_ + hd;
      const float v = acc[ns][r];
      a32[idx] = v;
      a16[idx] = (_Float16)v;
    }
  }
}

// ---------------------------------------------------------------------------
// 6) Gate: g = attended@Wg[0:D] + trust*Wg[D] + bg; y = attended*sigmoid(g)
// ---------------------------------------------------------------------------
__global__ __launch_bounds__(256) void gate_gemm_kernel(
    const _Float16* __restrict__ A16, const float* __restrict__ a32,
    const float* __restrict__ Wg, const float* __restrict__ bg,
    const float* __restrict__ trust, _Float16* __restrict__ y16) {
  const int K = D_, N = D_;
  __shared__ _Float16 As[128 * 32];
  __shared__ _Float16 Bs[64 * 32];
  const int tile_m = blockIdx.x * 128;
  const int tile_n = blockIdx.y * 64;
  const int tid = threadIdx.x, lane = tid & 31, wave = tid >> 5;

  v8f acc[4] = {};
  const int arow = tid >> 1, ac0 = (tid & 1) * 16;
  const int bn = tid & 63, bk0 = (tid >> 6) * 8;

  for (int k0 = 0; k0 < K; k0 += 32) {
    const _Float16* ag = A16 + (size_t)(tile_m + arow) * K + k0 + ac0;
    copy16_g2l(&As[arow * 32 + ac0], ag);
    copy16_g2l(&As[arow * 32 + ac0 + 8], ag + 8);
    {
      const float* wg = Wg + (size_t)(k0 + bk0) * N + tile_n + bn;
      v8h hh;
#pragma unroll
      for (int i = 0; i < 8; ++i) hh[i] = (_Float16)wg[(size_t)i * N];
      *(v8h*)&Bs[bn * 32 + bk0] = hh;
    }
    async_join();
    __syncthreads();

    const int ak = (lane < 16) ? 0 : 8;
    const int mr = wave * 16 + (lane & 15);
    v16h a = frag2(&As[mr * 32 + ak], &As[mr * 32 + ak + 16]);
    const int bk = (lane < 16) ? 0 : 8;
#pragma unroll
    for (int ns = 0; ns < 4; ++ns) {
      const int nr = ns * 16 + (lane & 15);
      v16h b = frag2(&Bs[nr * 32 + bk], &Bs[nr * 32 + bk + 16]);
      acc[ns] = wmma_f16(a, b, acc[ns]);
    }
    __syncthreads();
  }

  const int roff = (lane < 16) ? 0 : 8;
#pragma unroll
  for (int ns = 0; ns < 4; ++ns) {
    const int n = tile_n + ns * 16 + (lane & 15);
    const float wlast = Wg[(size_t)D_ * D_ + n];  // row D of (D+1,N)
    const float bn2 = bg[n];
#pragma unroll
    for (int r = 0; r < 8; ++r) {
      const int m = tile_m + wave * 16 + roff + r;
      const float g = acc[ns][r] + bn2 + trust[m] * wlast;
      const size_t idx = (size_t)m * D_ + n;
      y16[idx] = (_Float16)(a32[idx] * sigmoidf_(g));
    }
  }
}

// ---------------------------------------------------------------------------
// 7) Output projection: out = y @ Wo + bo  (fp32 out)
// ---------------------------------------------------------------------------
__global__ __launch_bounds__(256) void out_gemm_kernel(
    const _Float16* __restrict__ Y16, const float* __restrict__ Wo,
    const float* __restrict__ bo, float* __restrict__ out) {
  const int K = D_, N = D_;
  __shared__ _Float16 As[128 * 32];
  __shared__ _Float16 Bs[64 * 32];
  const int tile_m = blockIdx.x * 128;
  const int tile_n = blockIdx.y * 64;
  const int tid = threadIdx.x, lane = tid & 31, wave = tid >> 5;

  v8f acc[4] = {};
  const int arow = tid >> 1, ac0 = (tid & 1) * 16;
  const int bn = tid & 63, bk0 = (tid >> 6) * 8;

  for (int k0 = 0; k0 < K; k0 += 32) {
    const _Float16* ag = Y16 + (size_t)(tile_m + arow) * K + k0 + ac0;
    copy16_g2l(&As[arow * 32 + ac0], ag);
    copy16_g2l(&As[arow * 32 + ac0 + 8], ag + 8);
    {
      const float* wg = Wo + (size_t)(k0 + bk0) * N + tile_n + bn;
      v8h hh;
#pragma unroll
      for (int i = 0; i < 8; ++i) hh[i] = (_Float16)wg[(size_t)i * N];
      *(v8h*)&Bs[bn * 32 + bk0] = hh;
    }
    async_join();
    __syncthreads();

    const int ak = (lane < 16) ? 0 : 8;
    const int mr = wave * 16 + (lane & 15);
    v16h a = frag2(&As[mr * 32 + ak], &As[mr * 32 + ak + 16]);
    const int bk = (lane < 16) ? 0 : 8;
#pragma unroll
    for (int ns = 0; ns < 4; ++ns) {
      const int nr = ns * 16 + (lane & 15);
      v16h b = frag2(&Bs[nr * 32 + bk], &Bs[nr * 32 + bk + 16]);
      acc[ns] = wmma_f16(a, b, acc[ns]);
    }
    __syncthreads();
  }

  const int roff = (lane < 16) ? 0 : 8;
#pragma unroll
  for (int ns = 0; ns < 4; ++ns) {
    const int n = tile_n + ns * 16 + (lane & 15);
    const float bn2 = bo[n];
#pragma unroll
    for (int r = 0; r < 8; ++r) {
      const int m = tile_m + wave * 16 + roff + r;
      out[(size_t)m * D_ + n] = acc[ns][r] + bn2;
    }
  }
}

// ---------------------------------------------------------------------------
extern "C" void kernel_launch(void* const* d_in, const int* in_sizes, int n_in,
                              void* d_out, int out_size, void* d_ws, size_t ws_size,
                              hipStream_t stream) {
  (void)in_sizes; (void)n_in; (void)out_size; (void)ws_size;
  const float* x     = (const float*)d_in[0];
  const float* trust = (const float*)d_in[1];
  const float* Wq = (const float*)d_in[2];
  const float* bq = (const float*)d_in[3];
  const float* Wk = (const float*)d_in[4];
  const float* bk = (const float*)d_in[5];
  const float* Wv = (const float*)d_in[6];
  const float* bv = (const float*)d_in[7];
  const float* Wt = (const float*)d_in[8];
  const float* bt = (const float*)d_in[9];
  const float* Wg = (const float*)d_in[10];
  const float* bg = (const float*)d_in[11];
  const float* Wo = (const float*)d_in[12];
  const float* bo = (const float*)d_in[13];

  float* out  = (float*)d_out;
  float* attn = out + (size_t)M_ * D_;  // second tuple element, [B,H,S,S]

  _Float16* qh  = (_Float16*)d_ws;               // [B,H,S,HD] f16
  _Float16* kh  = qh  + (size_t)M_ * D_;         // [B,H,S,HD] f16
  _Float16* vhT = kh  + (size_t)M_ * D_;         // [B,H,HD,S] f16 (transposed)
  _Float16* tp  = vhT + (size_t)M_ * D_;         // [B,S,HD] f16
  _Float16* a16 = tp  + (size_t)M_ * HD_;        // attended f16 [B,S,D]
  _Float16* y16 = a16 + (size_t)M_ * D_;         // gated f16 [B,S,D]
  float*    a32 = (float*)(y16 + (size_t)M_ * D_);  // attended fp32 [B,S,D]

  dim3 gemm_grid(M_ / 128, D_ / 64);
  proj_gemm_kernel<<<gemm_grid, 256, 0, stream>>>(x, Wq, bq, qh, HD_, 1);
  proj_gemm_kernel<<<gemm_grid, 256, 0, stream>>>(x, Wk, bk, kh, HD_, 1);
  proj_gemm_kernel<<<gemm_grid, 256, 0, stream>>>(x, Wv, bv, vhT, 1, S_);

  trust_proj_kernel<<<(M_ * HD_) / 256, 256, 0, stream>>>(trust, Wt, bt, tp);

  dim3 sc_grid(S_ / 128, S_ / 64, BH_);
  scores_kernel<<<sc_grid, 256, 0, stream>>>(qh, kh, tp, attn);

  softmax_kernel<<<BH_ * S_, 256, 0, stream>>>(attn);

  dim3 av_grid(S_ / 128, BH_);
  av_gemm_kernel<<<av_grid, 256, 0, stream>>>(attn, vhT, a32, a16);

  gate_gemm_kernel<<<gemm_grid, 256, 0, stream>>>(a16, a32, Wg, bg, trust, y16);
  out_gemm_kernel<<<gemm_grid, 256, 0, stream>>>(y16, Wo, bo, out);
}
